// EGATWrapper_83330955477486
// MI455X (gfx1250) — matrix operand order repaired and checked by reading, not compiled, not run on hardware
//
#include <hip/hip_runtime.h>

typedef __attribute__((ext_vector_type(16))) _Float16 v16h;
typedef __attribute__((ext_vector_type(8)))  float    v8f;
typedef __attribute__((ext_vector_type(4)))  unsigned u32x4;
typedef __attribute__((ext_vector_type(8)))  int      i32x8;
typedef __attribute__((ext_vector_type(4)))  int      i32x4;

// ---------------------------------------------------------------------------
// Fragment loaders (wave32), using this code's own contiguous k order (same for A and B).
// A (16x32 f16): lane m = lane&15; K = p + (lane>=16 ? 16 : 0), contiguous in k
// B (32x16 f16): lane n = lane&15; K = p + (lane>=16 ? 16 : 0), contiguous in k
// C (16x16 f32): n = lane&15, m = r + (lane>=16 ? 8 : 0)
// ---------------------------------------------------------------------------
__device__ __forceinline__ v16h load_a_frag(const _Float16* row, int k0, int lane) {
  const int koff = (lane >= 16) ? 16 : 0;
  v16h a;
#pragma unroll
  for (int p = 0; p < 16; ++p) {
    const int K = p + koff;
    a[p] = row[k0 + K];
  }
  return a;
}

__device__ __forceinline__ v16h load_b_frag(const _Float16* W, int L, int n0, int k0, int lane) {
  const int n   = lane & 15;
  const int khl = (lane >= 16) ? 16 : 0;
  const _Float16* p = W + (n0 + n) * L + k0 + khl;
  v16h b;
#pragma unroll
  for (int q = 0; q < 16; ++q) b[q] = p[q];
  return b;
}

__device__ __forceinline__ v8f wmma_f16(v16h a, v16h b, v8f c) {
  return __builtin_amdgcn_wmma_f32_16x16x32_f16(false, a, false, b, (short)0, c, false, false);
}

// Pack 4 floats -> 4 f16 and store as 8B to (16B-aligned-array) LDS.
__device__ __forceinline__ void store4h(_Float16* dstp, float a, float b, float c, float d) {
  union { _Float16 h[4]; uint2 u; } cv;
  cv.h[0] = (_Float16)a; cv.h[1] = (_Float16)b;
  cv.h[2] = (_Float16)c; cv.h[3] = (_Float16)d;
  *(uint2*)dstp = cv.u;
}

// Set a bitfield inside a little-endian dword array (for TDM D# packing).
__device__ __forceinline__ void setbits(unsigned* w, int lo, int len, unsigned long long val) {
#pragma unroll 1
  for (int i = 0; i < len; ++i) {
    if ((val >> i) & 1ull) {
      const int bit = lo + i;
      w[bit >> 5] |= (1u << (bit & 31));
    }
  }
}

// ---------------------------------------------------------------------------
// Tiny prep: wsum[d] = sum_k Wattn[k,d]; wvec = Wep[:,64];
// ef_base[r,:] = rel_emb[r] @ Wep[:, :64]^T + bep
// ---------------------------------------------------------------------------
__global__ void k_prep(const float* __restrict__ rel_emb, const float* __restrict__ Wep,
                       const float* __restrict__ bep, const float* __restrict__ Wattn,
                       float* __restrict__ ef_base, float* __restrict__ wvec,
                       float* __restrict__ wsum, int n_rel) {
  const int tid = threadIdx.x;
  if (tid < 64) {
    float s = 0.f;
#pragma unroll
    for (int k = 0; k < 4; ++k) s += Wattn[k * 64 + tid];
    wsum[tid] = s;
    wvec[tid] = Wep[tid * 65 + 64];
  }
  for (int i = tid; i < n_rel * 64; i += blockDim.x) {
    const int r = i >> 6, o = i & 63;
    float s = bep[o];
#pragma unroll 8
    for (int j = 0; j < 64; ++j) s += rel_emb[r * 64 + j] * Wep[o * 65 + j];
    ef_base[i] = s;
  }
}

__global__ void k_init(float* __restrict__ xout, unsigned* __restrict__ amaxU,
                       float* __restrict__ denom, int N) {
  const int i = blockIdx.x * blockDim.x + threadIdx.x;
  if (i < N * 64) xout[i] = 0.f;
  if (i < N * 4) { amaxU[i] = 0u; denom[i] = 0.f; }
}

// ---------------------------------------------------------------------------
// x = emb[entity] @ Wp^T + bp   (N x 64, K = 128)
// ---------------------------------------------------------------------------
__global__ void k_node_x(const int* __restrict__ entity, const float* __restrict__ emb,
                         const float* __restrict__ Wp, const float* __restrict__ bp,
                         float* __restrict__ x, int N) {
  __shared__ __align__(16) _Float16 WpSh[64 * 128];       // 16 KB
  __shared__ __align__(16) _Float16 stage[4][16 * 128];   // 16 KB
  const int tid = threadIdx.x;
  for (int i = tid; i < 64 * 32; i += blockDim.x) {
    const float4 v = *(const float4*)&Wp[i * 4];
    store4h(&WpSh[i * 4], v.x, v.y, v.z, v.w);
  }
  __syncthreads();
  const int wave = tid >> 5, lane = tid & 31;
  const int ncol = lane & 15;
  const int mb = (lane >= 16) ? 8 : 0;
  const int ntiles = (N + 15) >> 4;
  for (int tile = blockIdx.x * 4 + wave; tile < ntiles; tile += gridDim.x * 4) {
    const int n0 = tile << 4;
    _Float16* st = stage[wave];
    for (int i = lane; i < 16 * 32; i += 32) {          // 4 feats per step
      const int m = i >> 5, k = (i & 31) * 4;
      int row = n0 + m; if (row >= N) row = N - 1;
      const float4 v = *(const float4*)&emb[(long long)entity[row] * 128 + k];
      store4h(&st[m * 128 + k], v.x, v.y, v.z, v.w);
    }
    asm volatile("s_wait_dscnt 0" ::: "memory");
    v16h afr[4];
    const _Float16* rb = st + ncol * 128;
#pragma unroll
    for (int kt = 0; kt < 4; ++kt) afr[kt] = load_a_frag(rb, kt * 32, lane);
    const bool full = (n0 + 16 <= N);
#pragma unroll
    for (int nt = 0; nt < 4; ++nt) {
      v8f c = {};
#pragma unroll
      for (int kt = 0; kt < 4; ++kt)
        c = wmma_f16(afr[kt], load_b_frag(WpSh, 128, nt * 16, kt * 32, lane), c);
      const int col = nt * 16 + ncol;
      const float bias = bp[col];
      if (full) {
#pragma unroll
        for (int r = 0; r < 8; ++r) x[(n0 + mb + r) * 64 + col] = c[r] + bias;
      } else {
#pragma unroll
        for (int r = 0; r < 8; ++r) {
          const int row = n0 + mb + r;
          if (row < N) x[row * 64 + col] = c[r] + bias;
        }
      }
    }
  }
}

// ---------------------------------------------------------------------------
// nproj = x @ Wn^T + bn   (N x 256, K = 64)
// ---------------------------------------------------------------------------
__global__ void k_nproj(const float* __restrict__ x, const float* __restrict__ Wn,
                        const float* __restrict__ bn, float* __restrict__ nproj, int N) {
  __shared__ __align__(16) _Float16 WnSh[256 * 64];       // 32 KB
  __shared__ __align__(16) _Float16 stage[4][16 * 64];    // 8 KB
  const int tid = threadIdx.x;
  for (int i = tid; i < 256 * 16; i += blockDim.x) {
    const float4 v = *(const float4*)&Wn[i * 4];
    store4h(&WnSh[i * 4], v.x, v.y, v.z, v.w);
  }
  __syncthreads();
  const int wave = tid >> 5, lane = tid & 31;
  const int ncol = lane & 15;
  const int mb = (lane >= 16) ? 8 : 0;
  const int ntiles = (N + 15) >> 4;
  for (int tile = blockIdx.x * 4 + wave; tile < ntiles; tile += gridDim.x * 4) {
    const int n0 = tile << 4;
    _Float16* st = stage[wave];
    for (int i = lane; i < 16 * 16; i += 32) {
      const int m = i >> 4, k = (i & 15) * 4;
      int row = n0 + m; if (row >= N) row = N - 1;
      const float4 v = *(const float4*)&x[row * 64 + k];
      store4h(&st[m * 64 + k], v.x, v.y, v.z, v.w);
    }
    asm volatile("s_wait_dscnt 0" ::: "memory");
    v16h afr[2];
    const _Float16* rb = st + ncol * 64;
#pragma unroll
    for (int kt = 0; kt < 2; ++kt) afr[kt] = load_a_frag(rb, kt * 32, lane);
    const bool full = (n0 + 16 <= N);
#pragma unroll
    for (int nt = 0; nt < 16; ++nt) {
      v8f c = {};
#pragma unroll
      for (int kt = 0; kt < 2; ++kt)
        c = wmma_f16(afr[kt], load_b_frag(WnSh, 64, nt * 16, kt * 32, lane), c);
      const int col = nt * 16 + ncol;
      const float bias = bn[col];
      if (full) {
#pragma unroll
        for (int r = 0; r < 8; ++r) nproj[(n0 + mb + r) * 256 + col] = c[r] + bias;
      } else {
#pragma unroll
        for (int r = 0; r < 8; ++r) {
          const int row = n0 + mb + r;
          if (row < N) nproj[row * 256 + col] = c[r] + bias;
        }
      }
    }
  }
}

// ---------------------------------------------------------------------------
// Edge GEMM: f_out = leaky_relu(stack @ We^T); a computed in-register.
// We (256x192 f32) is DMA'd into LDS by the Tensor Data Mover in 32-row
// chunks, then converted to f16. 96 v_wmma per 16-edge tile.
// LDS: 96 KB WeSh + 24 KB TDM stage + 24 KB stack stage = 144 KB.
// ---------------------------------------------------------------------------
__global__ void k_edge(const int* __restrict__ src, const int* __restrict__ dst,
                       const int* __restrict__ etype, const float* __restrict__ ew,
                       const float* __restrict__ x, const float* __restrict__ We,
                       const float* __restrict__ ef_base, const float* __restrict__ wvec,
                       const float* __restrict__ wsum,
                       float* __restrict__ f_out, float* __restrict__ a_out, int E) {
  __shared__ __align__(16) _Float16 WeSh[256 * 192];      // 96 KB
  __shared__ __align__(16) float    WeStage[32 * 192];    // 24 KB (TDM landing)
  __shared__ __align__(16) _Float16 stage[4][16 * 192];   // 24 KB
  const int tid = threadIdx.x;

  // --- Load We via Tensor Data Mover: 8 chunks of 32 rows x 192 cols (f32) ---
  for (int c = 0; c < 8; ++c) {
    if (tid < 32) {                                        // wave 0 issues TDM
      unsigned g0w[4] = {0u, 0u, 0u, 0u};
      unsigned g1w[8] = {0u, 0u, 0u, 0u, 0u, 0u, 0u, 0u};
      const unsigned long long ga =
          (unsigned long long)(const void*)(We + (size_t)c * 32 * 192);
      const unsigned lds_addr = (unsigned)(size_t)(void*)WeStage;
      // D# group 0: count=1, lds_addr, global_addr, type=2 (ISA 8.3)
      setbits(g0w, 0, 2, 1ull);          // count = 1
      setbits(g0w, 32, 32, lds_addr);    // lds_addr
      setbits(g0w, 64, 57, ga);          // global_addr
      setbits(g0w, 126, 2, 2ull);        // type = 2 ("image")
      // D# group 1: data_size=4B, 2D tensor 192x256, tile 192x32 (ISA 8.4)
      setbits(g1w, 16, 2, 2ull);         // data_size = 4 bytes
      setbits(g1w, 48, 32, 192ull);      // tensor_dim0
      setbits(g1w, 80, 32, 256ull);      // tensor_dim1
      setbits(g1w, 112, 16, 192ull);     // tile_dim0
      setbits(g1w, 128, 16, 32ull);      // tile_dim1
      setbits(g1w, 160, 48, 192ull);     // tensor_dim0_stride
      u32x4 g0; i32x8 g1; i32x4 gz; i32x8 gz8;
#pragma unroll
      for (int j = 0; j < 4; ++j) { g0[j] = g0w[j]; gz[j] = 0; }
#pragma unroll
      for (int j = 0; j < 8; ++j) { g1[j] = (int)g1w[j]; gz8[j] = 0; }
      __builtin_amdgcn_tensor_load_to_lds(g0, g1, gz, gz, gz8, 0);
      __builtin_amdgcn_s_wait_tensorcnt(0);
    }
    __syncthreads();
    for (int i = tid; i < 32 * 48; i += blockDim.x) {      // f32 -> f16, 4-wide
      const float4 v = *(const float4*)&WeStage[i * 4];
      store4h(&WeSh[c * 32 * 192 + i * 4], v.x, v.y, v.z, v.w);
    }
    __syncthreads();
  }

  const int wave = tid >> 5, lane = tid & 31;
  const int ncol = lane & 15;
  const int mb = (lane >= 16) ? 8 : 0;
  float wsv[4];
#pragma unroll
  for (int q = 0; q < 4; ++q) wsv[q] = wsum[q * 16 + ncol];
  const int tiles = E >> 4;
  for (int tile = blockIdx.x * 4 + wave; tile < tiles; tile += gridDim.x * 4) {
    const int e0 = tile << 4;
    int s = 0, d = 0, t = 0; float w = 0.f;
    if (lane < 16) {
      s = src[e0 + lane]; d = dst[e0 + lane];
      t = etype[e0 + lane]; w = ew[e0 + lane];
    }
    _Float16* st = stage[wave];
    for (int i = lane; i < 16 * 48; i += 32) {             // 4 feats per step
      const int m = i / 48, k = (i - m * 48) * 4;
      const int   sm = __shfl(s, m, 32);
      const int   dm = __shfl(d, m, 32);
      const int   tm = __shfl(t, m, 32);
      const float wm = __shfl(w, m, 32);
      float4 v;
      if (k < 64) {
        v = *(const float4*)&x[sm * 64 + k];
      } else if (k < 128) {
        const int o = k - 64;
        const float4 eb = *(const float4*)&ef_base[tm * 64 + o];
        const float4 wv = *(const float4*)&wvec[o];
        v.x = eb.x + wm * wv.x; v.y = eb.y + wm * wv.y;
        v.z = eb.z + wm * wv.z; v.w = eb.w + wm * wv.w;
      } else {
        v = *(const float4*)&x[dm * 64 + (k - 128)];
      }
      store4h(&st[m * 192 + k], v.x, v.y, v.z, v.w);
    }
    asm volatile("s_wait_dscnt 0" ::: "memory");
    v16h afr[6];
    const _Float16* rb = st + ncol * 192;
#pragma unroll
    for (int kt = 0; kt < 6; ++kt) afr[kt] = load_a_frag(rb, kt * 32, lane);
    float aacc[8];
#pragma unroll
    for (int r = 0; r < 8; ++r) aacc[r] = 0.f;
#pragma unroll
    for (int nt = 0; nt < 16; ++nt) {
      v8f c = {};
#pragma unroll
      for (int kt = 0; kt < 6; ++kt)
        c = wmma_f16(afr[kt], load_b_frag(WeSh, 192, nt * 16, kt * 32, lane), c);
      const float wsq = wsv[nt & 3];
#pragma unroll
      for (int r = 0; r < 8; ++r) {
        float v = c[r];
        v = (v > 0.f) ? v : v * 0.01f;                     // leaky_relu slope 0.01
        f_out[(long long)(e0 + mb + r) * 256 + nt * 16 + ncol] = v;
        aacc[r] += v * wsq;
      }
      if ((nt & 3) == 3) {                                 // head complete: reduce a
        const int h = nt >> 2;
#pragma unroll
        for (int r = 0; r < 8; ++r) {
          float v = aacc[r];
          v += __shfl_xor(v, 1, 32);
          v += __shfl_xor(v, 2, 32);
          v += __shfl_xor(v, 4, 32);
          v += __shfl_xor(v, 8, 32);
          if (ncol == 0) a_out[(e0 + mb + r) * 4 + h] = v;
          aacc[r] = 0.f;
        }
      }
    }
  }
}

// ---------------------------------------------------------------------------
// Segment softmax + aggregation (order-preserving uint encoding for max)
// ---------------------------------------------------------------------------
__device__ __forceinline__ unsigned f2key(float f) {
  unsigned u = __float_as_uint(f);
  return (u & 0x80000000u) ? ~u : (u | 0x80000000u);
}
__device__ __forceinline__ float key2f(unsigned key) {
  unsigned u = (key & 0x80000000u) ? (key & 0x7fffffffu) : ~key;
  return __uint_as_float(u);
}

__global__ void k_amax(const int* __restrict__ dst, const float* __restrict__ a,
                       unsigned* __restrict__ amaxU, int E) {
  const int i = blockIdx.x * blockDim.x + threadIdx.x;
  if (i >= E * 4) return;
  const int e = i >> 2, h = i & 3;
  atomicMax(&amaxU[dst[e] * 4 + h], f2key(a[i]));
}

__global__ void k_ex(const int* __restrict__ dst, const float* __restrict__ a,
                     const unsigned* __restrict__ amaxU, float* __restrict__ ex,
                     float* __restrict__ denom, int E) {
  const int i = blockIdx.x * blockDim.x + threadIdx.x;
  if (i >= E * 4) return;
  const int e = i >> 2, h = i & 3;
  const float amax = key2f(amaxU[dst[e] * 4 + h]);
  const float v = __expf(a[i] - amax);
  ex[i] = v;
  atomicAdd(&denom[dst[e] * 4 + h], v);
}

// alpha normalization in place: ex[i] /= denom[dst, h]  (one divide per (e,h))
__global__ void k_alpha(const int* __restrict__ dst, const float* __restrict__ denom,
                        float* __restrict__ ex, int E) {
  const int i = blockIdx.x * blockDim.x + threadIdx.x;
  if (i >= E * 4) return;
  const int e = i >> 2, h = i & 3;
  ex[i] = ex[i] / denom[dst[e] * 4 + h];
}

// x_out[dst] += 0.25 * sum_h alpha[e,h] * nproj[src, h, :]   (float4 per thread)
__global__ void k_agg(const int* __restrict__ src, const int* __restrict__ dst,
                      const float* __restrict__ alpha, const float* __restrict__ nproj,
                      float* __restrict__ xout, int E) {
  const long long i = (long long)blockIdx.x * blockDim.x + threadIdx.x;
  if (i >= (long long)E * 16) return;
  const int e = (int)(i >> 4), dc4 = (int)(i & 15) * 4;
  const int s = src[e], d = dst[e];
  float ax = 0.f, ay = 0.f, az = 0.f, aw = 0.f;
#pragma unroll
  for (int h = 0; h < 4; ++h) {
    const float al = alpha[e * 4 + h];
    const float4 np = *(const float4*)&nproj[s * 256 + h * 64 + dc4];
    ax += al * np.x; ay += al * np.y; az += al * np.z; aw += al * np.w;
  }
  float* o = &xout[d * 64 + dc4];
  atomicAdd(o + 0, 0.25f * ax);
  atomicAdd(o + 1, 0.25f * ay);
  atomicAdd(o + 2, 0.25f * az);
  atomicAdd(o + 3, 0.25f * aw);
}

// ---------------------------------------------------------------------------
extern "C" void kernel_launch(void* const* d_in, const int* in_sizes, int n_in,
                              void* d_out, int out_size, void* d_ws, size_t ws_size,
                              hipStream_t stream) {
  const int*   entity = (const int*)d_in[0];
  const int*   eidx   = (const int*)d_in[1];
  const int*   etype  = (const int*)d_in[2];
  const float* ew     = (const float*)d_in[3];
  const float* emb    = (const float*)d_in[4];
  const float* Wp     = (const float*)d_in[5];
  const float* bp     = (const float*)d_in[6];
  const float* rel    = (const float*)d_in[7];
  const float* Wep    = (const float*)d_in[8];
  const float* bep    = (const float*)d_in[9];
  const float* Wn     = (const float*)d_in[10];
  const float* bn     = (const float*)d_in[11];
  const float* We     = (const float*)d_in[12];
  const float* Wattn  = (const float*)d_in[13];

  const int N = in_sizes[0];
  const int E = in_sizes[2];
  const int n_rel = in_sizes[7] / 64;

  float* xout = (float*)d_out;
  float* fout = (float*)d_out + (size_t)N * 64;

  float*    ws      = (float*)d_ws;
  float*    x       = ws;                               // N*64
  float*    nproj   = x + (size_t)N * 64;               // N*256
  float*    a       = nproj + (size_t)N * 256;          // E*4
  float*    ex      = a + (size_t)E * 4;                // E*4
  unsigned* amaxU   = (unsigned*)(ex + (size_t)E * 4);  // N*4
  float*    denom   = (float*)(amaxU + (size_t)N * 4);  // N*4
  float*    wsum    = denom + (size_t)N * 4;            // 64
  float*    wvec    = wsum + 64;                        // 64
  float*    ef_base = wvec + 64;                        // n_rel*64

  const int* srcp = eidx;
  const int* dstp = eidx + E;

  k_prep<<<1, 256, 0, stream>>>(rel, Wep, bep, Wattn, ef_base, wvec, wsum, n_rel);
  k_init<<<(N * 64 + 255) / 256, 256, 0, stream>>>(xout, amaxU, denom, N);

  const int ntiles  = (N + 15) / 16;
  const int nblocks = (ntiles + 3) / 4;
  k_node_x<<<nblocks, 128, 0, stream>>>(entity, emb, Wp, bp, x, N);
  k_nproj<<<nblocks, 128, 0, stream>>>(x, Wn, bn, nproj, N);

  const int etiles = E / 16;
  int eblocks = 2560;
  if (eblocks * 4 > etiles) eblocks = (etiles + 3) / 4;
  k_edge<<<eblocks, 128, 0, stream>>>(srcp, dstp, etype, ew, x, We, ef_base, wvec, wsum,
                                      fout, a, E);

  const long long eh = (long long)E * 4;
  k_amax<<<(int)((eh + 255) / 256), 256, 0, stream>>>(dstp, a, amaxU, E);
  k_ex<<<(int)((eh + 255) / 256), 256, 0, stream>>>(dstp, a, amaxU, ex, denom, E);
  k_alpha<<<(int)((eh + 255) / 256), 256, 0, stream>>>(dstp, denom, ex, E);

  const long long agg = (long long)E * 16;
  k_agg<<<(int)((agg + 255) / 256), 256, 0, stream>>>(srcp, dstp, ex, nproj, xout, E);
}
